// TransformerBlock_16561393893799
// MI455X (gfx1250) — matrix operand order repaired and checked by reading, not compiled, not run on hardware
//
#include <hip/hip_runtime.h>

// ---------------------------------------------------------------------------
// Transformer block for MI455X (gfx1250, wave32, WMMA).
// All GEMMs run through v_wmma_f32_16x16x32_bf16 (fp32 accumulate).
// ~77 GFLOP of GEMM work vs ~0.5 GB forced HBM traffic (the reference returns
// the 256MB attention tensor), so bf16 WMMA keeps math under the ~21us memory
// floor. Weights are transposed+converted to bf16 once per launch so every
// weight GEMM stages LDS tiles with contiguous 32B copies, which lets us use
// GLOBAL_LOAD_ASYNC_TO_LDS_B128 (ASYNCcnt) to overlap staging with WMMA math.
// ---------------------------------------------------------------------------

typedef unsigned short u16;
typedef __attribute__((ext_vector_type(16))) __bf16 bf16x16;
typedef __attribute__((ext_vector_type(8)))  float  f32x8;
typedef __attribute__((ext_vector_type(4)))  float  f32x4;
typedef __attribute__((ext_vector_type(4)))  unsigned int u32x4;
typedef int v4i __attribute__((vector_size(16)));   // matches async-LDS builtin pointee

#define S_LEN   2048
#define DMODEL  1024
#define DFF     4096
#define NHEADS  16
#define DHEAD   64
#define LN_EPS  1e-6f

// GEMM tiling: 128x128 block tile, BK=32, 256 threads = 8 waves (4x2),
// each wave owns a 32x64 sub-tile = 2x4 WMMA fragments.
#define BM 128
#define BN 128
#define BK 32
#define LDSTR 40   // BK + 8 pad: 80B row stride keeps 16B alignment, no bank conflicts

// ---- CDNA5 async global->LDS copy (guarded; falls back to sync staging) ----
#if defined(__gfx1250__) && __has_builtin(__builtin_amdgcn_global_load_async_to_lds_b128)
#define ASYNC_COPY 1
#else
#define ASYNC_COPY 0
#endif

__device__ __forceinline__ void async_ld_b128(const void* g, void* l) {
#if ASYNC_COPY
  __builtin_amdgcn_global_load_async_to_lds_b128(
      (__attribute__((address_space(1))) v4i*)g,
      (__attribute__((address_space(3))) v4i*)l, 0, 0);
#endif
}
__device__ __forceinline__ void wait_async0() {
#if ASYNC_COPY
#if __has_builtin(__builtin_amdgcn_s_wait_asynccnt)
  __builtin_amdgcn_s_wait_asynccnt(0);
#else
  asm volatile("s_wait_asynccnt 0x0" ::: "memory");
#endif
#endif
}

__device__ __forceinline__ u16 f32_to_bf16(float f) {
  unsigned u = __float_as_uint(f);
  if ((u & 0x7fffffffu) > 0x7f800000u) return (u16)((u >> 16) | 0x0040u); // quiet NaN
  u += 0x7fffu + ((u >> 16) & 1u);                                       // RNE
  return (u16)(u >> 16);
}

union Frag   { u32x4 q[2]; bf16x16 v; };
union Pack16 { u32x4 q[2]; u16 h[16]; };

// ---------------------------------------------------------------------------
// Block reductions (wave32)
// ---------------------------------------------------------------------------
__device__ __forceinline__ float warpSum(float v) {
#pragma unroll
  for (int o = 16; o > 0; o >>= 1) v += __shfl_xor(v, o, 32);
  return v;
}
__device__ __forceinline__ float warpMax(float v) {
#pragma unroll
  for (int o = 16; o > 0; o >>= 1) v = fmaxf(v, __shfl_xor(v, o, 32));
  return v;
}
__device__ __forceinline__ float blockSum(float v) {
  __shared__ float sm[8];
  v = warpSum(v);
  if ((threadIdx.x & 31) == 0) sm[threadIdx.x >> 5] = v;
  __syncthreads();
  if (threadIdx.x < 32) {
    float t = (threadIdx.x < 8) ? sm[threadIdx.x] : 0.f;
#pragma unroll
    for (int o = 4; o > 0; o >>= 1) t += __shfl_xor(t, o, 32);
    if (threadIdx.x == 0) sm[0] = t;
  }
  __syncthreads();
  float r = sm[0];
  __syncthreads();
  return r;
}
__device__ __forceinline__ float blockMax(float v) {
  __shared__ float sm[8];
  v = warpMax(v);
  if ((threadIdx.x & 31) == 0) sm[threadIdx.x >> 5] = v;
  __syncthreads();
  if (threadIdx.x < 32) {
    float t = (threadIdx.x < 8) ? sm[threadIdx.x] : -__builtin_inff();
#pragma unroll
    for (int o = 4; o > 0; o >>= 1) t = fmaxf(t, __shfl_xor(t, o, 32));
    if (threadIdx.x == 0) sm[0] = t;
  }
  __syncthreads();
  float r = sm[0];
  __syncthreads();
  return r;
}

// ---------------------------------------------------------------------------
// Transpose + fp32->bf16: dst[N][K] = bf16(src[K][N]); 64x64 LDS tiles.
// Run once per launch on the 8 weight matrices (~28MB total).
// ---------------------------------------------------------------------------
__global__ void transpose_cvt_bf16(const float* __restrict__ s, u16* __restrict__ d,
                                   int K, int N) {
  __shared__ __align__(16) u16 t[64][72];   // +8 pad
  const int k0 = blockIdx.y * 64, n0 = blockIdx.x * 64;
  {
    const int r  = threadIdx.x >> 2;          // src row in tile 0..63
    const int cs = (threadIdx.x & 3) * 16;    // src col seg
    const f32x4* g = (const f32x4*)(s + (size_t)(k0 + r) * N + n0 + cs);
#pragma unroll
    for (int i = 0; i < 4; i++) {
      f32x4 f = g[i];
      t[cs + i * 4 + 0][r] = f32_to_bf16(f.x);
      t[cs + i * 4 + 1][r] = f32_to_bf16(f.y);
      t[cs + i * 4 + 2][r] = f32_to_bf16(f.z);
      t[cs + i * 4 + 3][r] = f32_to_bf16(f.w);
    }
  }
  __syncthreads();
  {
    const int n  = threadIdx.x >> 2;          // dst row in tile 0..63
    const int ks = (threadIdx.x & 3) * 16;    // dst col seg
    u32x4* o = (u32x4*)(d + (size_t)(n0 + n) * K + k0 + ks);
    const u32x4* p = (const u32x4*)&t[n][ks];
    o[0] = p[0];
    o[1] = p[1];
  }
}

// ---------------------------------------------------------------------------
// LayerNorm: out = gamma*((x-mean)/(std+eps)) + beta  (std = sqrt(biased var))
// ---------------------------------------------------------------------------
__global__ void layernorm_kernel(const float* __restrict__ x,
                                 const float* __restrict__ gamma,
                                 const float* __restrict__ beta,
                                 float* __restrict__ of32,
                                 u16*   __restrict__ obf16,
                                 int D) {
  const size_t base = (size_t)blockIdx.x * D;
  const float* xr = x + base;
  float s = 0.f;
  for (int i = threadIdx.x; i < D; i += 256) s += xr[i];
  float mean = blockSum(s) / (float)D;
  float v = 0.f;
  for (int i = threadIdx.x; i < D; i += 256) { float d = xr[i] - mean; v += d * d; }
  float var = blockSum(v) / (float)D;
  float inv = 1.f / (sqrtf(var) + LN_EPS);
  for (int i = threadIdx.x; i < D; i += 256) {
    float y = gamma[i] * ((xr[i] - mean) * inv) + beta[i];
    if (of32)  of32[base + i]  = y;
    if (obf16) obf16[base + i] = f32_to_bf16(y);
  }
}

// ---------------------------------------------------------------------------
// Row softmax in place (masked entries hold -inf -> become exact 0)
// ---------------------------------------------------------------------------
__global__ void softmax_rows(float* __restrict__ a, int ncol) {
  float* r = a + (size_t)blockIdx.x * ncol;
  float m = -__builtin_inff();
  for (int i = threadIdx.x; i < ncol; i += 256) m = fmaxf(m, r[i]);
  m = blockMax(m);
  float s = 0.f;
  for (int i = threadIdx.x; i < ncol; i += 256) {
    float e = __expf(r[i] - m);
    r[i] = e;
    s += e;
  }
  s = blockSum(s);
  float inv = 1.f / s;
  for (int i = threadIdx.x; i < ncol; i += 256) r[i] *= inv;
}

// ---------------------------------------------------------------------------
// Batched WMMA GEMM (compile-time variant):  C[z] = epilogue( A[z] @ B[z] )
//   AF32=0: A is bf16 row-major; AF32=1: A fp32, converted while staging.
//   TRANSB=0: B bf16 row-major [K][N]; TRANSB=1: B bf16 [N][K] (ldb over N).
//   Epilogue: v = acc*scale + bias[col] (+resid[row,col]); relu; causal -inf.
//   fp32 and/or bf16 stores. Requires M%128==0, K%32==0, N%16==0.
// ---------------------------------------------------------------------------
template <bool AF32, bool TRANSB>
__global__ __launch_bounds__(256)
void gemm_wmma(int M, int N, int K,
               const void* __restrict__ Aptr, int lda, long long aBatch,
               const u16* __restrict__ B, int ldb, long long bBatch,
               float* __restrict__ Cf32, u16* __restrict__ Cbf,
               int ldc, long long cBatch,
               const float* __restrict__ bias,
               const float* __restrict__ resid,
               float scale, int causal, int relu) {
  __shared__ __align__(16) u16 As[2][BM * LDSTR];
  __shared__ __align__(16) u16 Bs[2][BN * LDSTR];

  constexpr bool kAsyncA = !AF32;
  constexpr bool kAsyncB = TRANSB;
  constexpr bool kAnyAsync = (kAsyncA || kAsyncB) && ASYNC_COPY;

  const int tid   = threadIdx.x;
  const int tileM = blockIdx.y * BM;
  const int tileN = blockIdx.x * BN;
  const int z     = blockIdx.z;

  float* Cg = Cf32 ? Cf32 + z * cBatch : nullptr;
  u16*   Cb = Cbf  ? Cbf  + z * cBatch : nullptr;

  // Fully-masked causal tile: write -inf, skip all math (uniform per block).
  if (causal && tileN > tileM + (BM - 1)) {
    for (int i = tid; i < BM * BN; i += 256) {
      int r = i >> 7, c = i & (BN - 1);
      int col = tileN + c;
      if (col < N) {
        size_t off = (size_t)(tileM + r) * ldc + col;
        if (Cg) Cg[off] = -__builtin_inff();
        if (Cb) Cb[off] = f32_to_bf16(-__builtin_inff());
      }
    }
    return;
  }

  const u16*   Ab = AF32 ? nullptr : (const u16*)Aptr + z * aBatch;
  const float* Af = AF32 ? (const float*)Aptr + z * aBatch : nullptr;
  const u16*   Bb = B + z * bBatch;

  // staging assignments (256 threads)
  const int arow = tid >> 1;           // 0..127
  const int aseg = (tid & 1) * 16;     // K sub-segment 0 / 16
  const int bkk  = tid >> 3;           // 0..31   (TRANSB=0)
  const int bseg = (tid & 7) * 16;     // col 0..112
  const int brow = tid >> 1;           // 0..127  (TRANSB=1: n)
  const int bks  = (tid & 1) * 16;     // K sub-segment 0 / 16

  auto stage = [&](int buf, int k0) {
    // ---- A tile: BM x BK, K-contiguous in LDS ----
    u16* adst = &As[buf][arow * LDSTR + aseg];
    if constexpr (!AF32) {
      const u16* g = Ab + (size_t)(tileM + arow) * lda + k0 + aseg;
#if ASYNC_COPY
      async_ld_b128(g, adst);
      async_ld_b128(g + 8, adst + 8);
#else
      const u32x4* gv = (const u32x4*)g;
      *(u32x4*)(adst)     = gv[0];
      *(u32x4*)(adst + 8) = gv[1];
#endif
      if (k0 + 2 * BK < K)  // warm L2/WGP$ for the tile after next
        __builtin_prefetch(g + 2 * BK, 0, 1);
    } else {
      const f32x4* g = (const f32x4*)(Af + (size_t)(tileM + arow) * lda + k0 + aseg);
      Pack16 p;
#pragma unroll
      for (int i = 0; i < 4; i++) {
        f32x4 f = g[i];
        p.h[i * 4 + 0] = f32_to_bf16(f.x);
        p.h[i * 4 + 1] = f32_to_bf16(f.y);
        p.h[i * 4 + 2] = f32_to_bf16(f.z);
        p.h[i * 4 + 3] = f32_to_bf16(f.w);
      }
      *(u32x4*)(adst)     = p.q[0];
      *(u32x4*)(adst + 8) = p.q[1];
    }
    // ---- B tile, stored [n][k] (K-contiguous) in LDS ----
    if constexpr (TRANSB) {
      u16* bdst = &Bs[buf][brow * LDSTR + bks];
      if (tileN + brow < N) {
        const u16* g = Bb + (size_t)(tileN + brow) * ldb + k0 + bks;
#if ASYNC_COPY
        async_ld_b128(g, bdst);
#else
        *(u32x4*)(bdst) = *(const u32x4*)g;
#endif
      } else {
        *(u32x4*)(bdst) = (u32x4){0, 0, 0, 0};
      }
    } else {
      Pack16 p;
      if (tileN + bseg < N) {
        const u32x4* g = (const u32x4*)(Bb + (size_t)(k0 + bkk) * ldb + tileN + bseg);
        p.q[0] = g[0]; p.q[1] = g[1];
      } else {
        p.q[0] = (u32x4){0, 0, 0, 0}; p.q[1] = (u32x4){0, 0, 0, 0};
      }
#pragma unroll
      for (int i = 0; i < 16; i++) Bs[buf][(bseg + i) * LDSTR + bkk] = p.h[i];
    }
  };

  // wave layout
  const int lane = tid & 31;
  const int wid  = tid >> 5;
  const int wm   = (wid & 3) * 32;   // wave row origin in tile
  const int wn   = (wid >> 2) * 64;  // wave col origin in tile
  const int l15  = lane & 15;
  const int ak   = (lane < 16) ? 0 : 16;  // A frag: lanes 0-15 K 0..15, lanes 16-31 K 16..31
  const int bk2  = (lane < 16) ? 0 : 16;  // B frag: lanes 0-15 K 0..15, lanes 16-31 K 16..31

  f32x8 acc[2][4] = {};

  stage(0, 0);
  if constexpr (kAnyAsync) wait_async0();
  __syncthreads();
  int buf = 0;
  for (int k0 = 0; k0 < K; k0 += BK) {
    if (k0 + BK < K) stage(buf ^ 1, k0 + BK);   // overlaps the 8 WMMAs below

    Frag a[2], b[4];
#pragma unroll
    for (int fr = 0; fr < 2; fr++) {
      const u16* p = &As[buf][(wm + fr * 16 + l15) * LDSTR + ak];
      a[fr].q[0] = *(const u32x4*)(p);
      a[fr].q[1] = *(const u32x4*)(p + 8);
    }
#pragma unroll
    for (int fc = 0; fc < 4; fc++) {
      const u16* p = &Bs[buf][(wn + fc * 16 + l15) * LDSTR + bk2];
      b[fc].q[0] = *(const u32x4*)(p);
      b[fc].q[1] = *(const u32x4*)(p + 8);
    }
#pragma unroll
    for (int fr = 0; fr < 2; fr++)
#pragma unroll
      for (int fc = 0; fc < 4; fc++)
        acc[fr][fc] = __builtin_amdgcn_wmma_f32_16x16x32_bf16(
            false, a[fr].v, false, b[fc].v, (short)0, acc[fr][fc], false, false);

    if constexpr (kAnyAsync) wait_async0();     // own buf^1 asyncs landed
    __syncthreads();                            // everyone's landed
    buf ^= 1;
  }

  // epilogue: C layout = VGPR r -> row (r + 8*(lane>=16)), col = lane&15
#pragma unroll
  for (int fr = 0; fr < 2; fr++) {
#pragma unroll
    for (int fc = 0; fc < 4; fc++) {
      int col = tileN + wn + fc * 16 + l15;
      if (col >= N) continue;
      int rbase = tileM + wm + fr * 16 + ((lane >> 4) << 3);
      float bc = bias ? bias[col] : 0.f;
#pragma unroll
      for (int r = 0; r < 8; r++) {
        int row = rbase + r;
        float v = acc[fr][fc][r] * scale + bc;
        if (resid) v += resid[(size_t)row * ldc + col];
        if (relu)  v = fmaxf(v, 0.f);
        if (causal && col > row) v = -__builtin_inff();
        size_t off = (size_t)row * ldc + col;
        if (Cg) Cg[off] = v;
        if (Cb) Cb[off] = f32_to_bf16(v);
      }
    }
  }
}

// ---------------------------------------------------------------------------
// Host side
// ---------------------------------------------------------------------------
static inline void launch_gemm(hipStream_t s, bool af32, bool transb,
                               int M, int N, int K,
                               const void* A, int lda, long long aB,
                               const u16* B, int ldb, long long bB, int batches,
                               float* Cf32, u16* Cbf, int ldc, long long cB,
                               const float* bias, const float* resid,
                               float scale, int causal, int relu) {
  dim3 grid((N + BN - 1) / BN, (M + BM - 1) / BM, batches);
  if (!af32 && transb)
    gemm_wmma<false, true><<<grid, 256, 0, s>>>(M, N, K, A, lda, aB, B, ldb, bB,
                                                Cf32, Cbf, ldc, cB, bias, resid, scale, causal, relu);
  else if (af32 && !transb)
    gemm_wmma<true, false><<<grid, 256, 0, s>>>(M, N, K, A, lda, aB, B, ldb, bB,
                                                Cf32, Cbf, ldc, cB, bias, resid, scale, causal, relu);
  else if (!af32 && !transb)
    gemm_wmma<false, false><<<grid, 256, 0, s>>>(M, N, K, A, lda, aB, B, ldb, bB,
                                                 Cf32, Cbf, ldc, cB, bias, resid, scale, causal, relu);
  else
    gemm_wmma<true, true><<<grid, 256, 0, s>>>(M, N, K, A, lda, aB, B, ldb, bB,
                                               Cf32, Cbf, ldc, cB, bias, resid, scale, causal, relu);
}

extern "C" void kernel_launch(void* const* d_in, const int* in_sizes, int n_in,
                              void* d_out, int out_size, void* d_ws, size_t ws_size,
                              hipStream_t stream) {
  const float* x    = (const float*)d_in[0];
  const float* g1   = (const float*)d_in[1];
  const float* be1  = (const float*)d_in[2];
  const float* W_in = (const float*)d_in[3];
  const float* b_in = (const float*)d_in[4];
  const float* Wq   = (const float*)d_in[5];
  const float* bq   = (const float*)d_in[6];
  const float* Wk   = (const float*)d_in[7];
  const float* bkv  = (const float*)d_in[8];
  const float* Wv   = (const float*)d_in[9];
  const float* bv   = (const float*)d_in[10];
  const float* Wo   = (const float*)d_in[11];
  const float* bo   = (const float*)d_in[12];
  const float* W2   = (const float*)d_in[13];
  const float* b2   = (const float*)d_in[14];
  const float* g2   = (const float*)d_in[15];
  const float* be2  = (const float*)d_in[16];
  const float* Wf1  = (const float*)d_in[17];
  const float* bf1  = (const float*)d_in[18];
  const float* Wf2  = (const float*)d_in[19];
  const float* bf2  = (const float*)d_in[20];

  float* out  = (float*)d_out;                          // [2048,1024]
  float* attn = out + (size_t)S_LEN * DMODEL;           // [16,2048,2048]

  // workspace carve (~92 MB)
  char* w = (char*)d_ws;
  auto carve = [&](size_t bytes) -> char* {
    char* p = w; w += (bytes + 255) & ~(size_t)255; return p;
  };
  // transposed bf16 weights: Wt[N][K]
  u16* W_in_t = (u16*)carve(sizeof(u16) * DMODEL * DMODEL);
  u16* Wq_t   = (u16*)carve(sizeof(u16) * DMODEL * DMODEL);
  u16* Wk_t   = (u16*)carve(sizeof(u16) * DMODEL * DMODEL);
  u16* Wv_t   = (u16*)carve(sizeof(u16) * DMODEL * DMODEL);
  u16* Wo_t   = (u16*)carve(sizeof(u16) * DMODEL * DMODEL);
  u16* W2_t   = (u16*)carve(sizeof(u16) * DMODEL * DMODEL);
  u16* Wf1_t  = (u16*)carve(sizeof(u16) * DMODEL * DFF);
  u16* Wf2_t  = (u16*)carve(sizeof(u16) * DFF * DMODEL);
  u16* ln1_b  = (u16*)carve(sizeof(u16) * S_LEN * DMODEL);
  u16* h_b    = (u16*)carve(sizeof(u16) * S_LEN * DMODEL);
  u16* q_b    = (u16*)carve(sizeof(u16) * S_LEN * DMODEL);
  u16* k_b    = (u16*)carve(sizeof(u16) * S_LEN * DMODEL);
  u16* v_b    = (u16*)carve(sizeof(u16) * S_LEN * DMODEL);
  u16* ctx_b  = (u16*)carve(sizeof(u16) * S_LEN * DMODEL);
  u16* mha_b  = (u16*)carve(sizeof(u16) * S_LEN * DMODEL);
  u16* n2_b   = (u16*)carve(sizeof(u16) * S_LEN * DMODEL);
  u16* ffh_b  = (u16*)carve(sizeof(u16) * S_LEN * DFF);
  float* part1 = (float*)carve(sizeof(float) * S_LEN * DMODEL);
  float* norm2 = (float*)carve(sizeof(float) * S_LEN * DMODEL);
  (void)ws_size; (void)in_sizes; (void)n_in; (void)out_size;

  // weight transpose+convert: src [K][N] -> dst [N][K] bf16
  auto tw = [&](const float* s, u16* d, int K, int N) {
    transpose_cvt_bf16<<<dim3(N / 64, K / 64), 256, 0, stream>>>(s, d, K, N);
  };
  tw(W_in, W_in_t, DMODEL, DMODEL);
  tw(Wq,   Wq_t,   DMODEL, DMODEL);
  tw(Wk,   Wk_t,   DMODEL, DMODEL);
  tw(Wv,   Wv_t,   DMODEL, DMODEL);
  tw(Wo,   Wo_t,   DMODEL, DMODEL);
  tw(W2,   W2_t,   DMODEL, DMODEL);
  tw(Wf1,  Wf1_t,  DMODEL, DFF);
  tw(Wf2,  Wf2_t,  DFF, DMODEL);

  // ln1
  layernorm_kernel<<<S_LEN, 256, 0, stream>>>(x, g1, be1, nullptr, ln1_b, DMODEL);
  // h = ln1 @ W_in + b_in            -> bf16
  launch_gemm(stream, false, true, S_LEN, DMODEL, DMODEL, ln1_b, DMODEL, 0,
              W_in_t, DMODEL, 0, 1, nullptr, h_b, DMODEL, 0, b_in, nullptr, 1.f, 0, 0);
  // q,k,v = h @ W{q,k,v} + b         -> bf16
  launch_gemm(stream, false, true, S_LEN, DMODEL, DMODEL, h_b, DMODEL, 0,
              Wq_t, DMODEL, 0, 1, nullptr, q_b, DMODEL, 0, bq, nullptr, 1.f, 0, 0);
  launch_gemm(stream, false, true, S_LEN, DMODEL, DMODEL, h_b, DMODEL, 0,
              Wk_t, DMODEL, 0, 1, nullptr, k_b, DMODEL, 0, bkv, nullptr, 1.f, 0, 0);
  launch_gemm(stream, false, true, S_LEN, DMODEL, DMODEL, h_b, DMODEL, 0,
              Wv_t, DMODEL, 0, 1, nullptr, v_b, DMODEL, 0, bv, nullptr, 1.f, 0, 0);
  // scores[h] = q_h @ k_h^T / 8, causal mask -> attn (fp32, part of d_out)
  launch_gemm(stream, false, true, S_LEN, S_LEN, DHEAD, q_b, DMODEL, DHEAD,
              k_b, DMODEL, DHEAD, NHEADS,
              attn, nullptr, S_LEN, (long long)S_LEN * S_LEN,
              nullptr, nullptr, 0.125f, 1, 0);
  // softmax rows (in place in d_out)
  softmax_rows<<<NHEADS * S_LEN, 256, 0, stream>>>(attn, S_LEN);
  // ctx[h] = attn_h @ v_h   (A = fp32 attn, converted while staging) -> bf16 [S, H*dk]
  launch_gemm(stream, true, false, S_LEN, DHEAD, S_LEN, attn, S_LEN, (long long)S_LEN * S_LEN,
              v_b, DMODEL, DHEAD, NHEADS,
              nullptr, ctx_b, DMODEL, DHEAD, nullptr, nullptr, 1.f, 0, 0);
  // mha = ctx @ Wo + bo              -> bf16
  launch_gemm(stream, false, true, S_LEN, DMODEL, DMODEL, ctx_b, DMODEL, 0,
              Wo_t, DMODEL, 0, 1, nullptr, mha_b, DMODEL, 0, bo, nullptr, 1.f, 0, 0);
  // part1 = x + mha @ W2 + b2        -> fp32
  launch_gemm(stream, false, true, S_LEN, DMODEL, DMODEL, mha_b, DMODEL, 0,
              W2_t, DMODEL, 0, 1, part1, nullptr, DMODEL, 0, b2, x, 1.f, 0, 0);
  // ln2 -> norm2 (fp32 for residual) + bf16 for FF
  layernorm_kernel<<<S_LEN, 256, 0, stream>>>(part1, g2, be2, norm2, n2_b, DMODEL);
  // ffh = relu(norm2 @ Wf1 + bf1)    -> bf16
  launch_gemm(stream, false, true, S_LEN, DFF, DMODEL, n2_b, DMODEL, 0,
              Wf1_t, DMODEL, 0, 1, nullptr, ffh_b, DFF, 0, bf1, nullptr, 1.f, 0, 1);
  // out = norm2 + ffh @ Wf2 + bf2    -> fp32 (d_out)
  launch_gemm(stream, false, true, S_LEN, DMODEL, DFF, ffh_b, DFF, 0,
              Wf2_t, DFF, 0, 1, out, nullptr, DMODEL, 0, bf2, norm2, 1.f, 0, 0);
}
